// Attention1_6597069766863
// MI455X (gfx1250) — matrix operand order repaired and checked by reading, not compiled, not run on hardware
//
#include <hip/hip_runtime.h>
#include <hip/hip_bf16.h>
#include <math.h>

// ---------------------------------------------------------------------------
// Attention: q,v from qx; k from hidden_states. H=2 heads, D=32, L=2048, B=8.
// All matmuls via v_wmma_f32_16x16x32_f16 (K=32 == HEAD_DIM -> 1 WMMA per
// 16x16 score tile). Flash-style 2-pass softmax (stats pass + recompute pass)
// to avoid staging 268MB of raw scores; probs written once, non-temporal
// (write-once stream must not thrash the 192MB L2). K/V/Q tiles staged into
// LDS via gfx1250 GLOBAL_LOAD_ASYNC_TO_LDS_B128 (ASYNCcnt-tracked DMA).
// ---------------------------------------------------------------------------

typedef __attribute__((ext_vector_type(16))) _Float16 v16h;
typedef __attribute__((ext_vector_type(8)))  _Float16 v8h;
typedef __attribute__((ext_vector_type(8)))  float    v8f;

constexpr int BATCH = 8;
constexpr int SEQ   = 2048;
constexpr int HID   = 64;
constexpr int NH    = 2;
constexpr int HD    = 32;
constexpr int KTILE = 128;   // keys per inner tile
constexpr int QTILE = 64;    // queries per workgroup (16 per wave)

// ---- gfx1250 async global->LDS staging (ASYNCcnt path) --------------------
#if defined(__gfx1250__) && __has_builtin(__builtin_amdgcn_global_load_async_to_lds_b128)
#define HAVE_ASYNC_LDS 1
#else
#define HAVE_ASYNC_LDS 0
#endif

// exact parameter types per clang diagnostic: int4 in AS1 (global) / AS3 (LDS)
typedef int v4i_ __attribute__((__vector_size__(16)));
typedef __attribute__((address_space(1))) v4i_ g_v4i;
typedef __attribute__((address_space(3))) v4i_ l_v4i;

__device__ __forceinline__ void cp16_lds(_Float16* dst, const _Float16* src) {
#if HAVE_ASYNC_LDS
  __builtin_amdgcn_global_load_async_to_lds_b128((g_v4i*)src, (l_v4i*)dst, 0, 0);
#else
  *(uint4*)dst = *(const uint4*)src;
#endif
}

__device__ __forceinline__ void wait_lds_cp() {
#if HAVE_ASYNC_LDS
#if __has_builtin(__builtin_amdgcn_s_wait_asynccnt)
  __builtin_amdgcn_s_wait_asynccnt(0);
#else
  asm volatile("s_wait_asynccnt 0" ::: "memory");
#endif
#endif
}

__device__ __forceinline__ v16h ld16pair(const _Float16* p0, const _Float16* p1) {
  v8h lo = *(const v8h*)p0;
  v8h hi = *(const v8h*)p1;
  return __builtin_shufflevector(lo, hi, 0,1,2,3,4,5,6,7,8,9,10,11,12,13,14,15);
}

// A operand (16x32 f16, row-major source M with leading dim ld):
// lanes 0-15: row m, K = {k0..k0+7, k0+16..k0+23}
// lanes 16-31: row m, K = {k0+8..k0+15, k0+24..k0+31}
__device__ __forceinline__ v16h load_A(const _Float16* M, int ld, int m0, int k0, int lane) {
  int m = m0 + (lane & 15);
  const _Float16* p = M + m * ld + k0 + ((lane >> 4) << 3);
  return ld16pair(p, p + 16);
}

// B operand (32x16 f16) where B[k][n] = M[n][k], M row-major (k contiguous):
// lanes 0-15: col n, K = {k0..k0+7, k0+16..k0+23} ; lanes 16-31: col n, K = {k0+8..k0+15, k0+24..k0+31}
__device__ __forceinline__ v16h load_Bt(const _Float16* M, int ld, int n0, int k0, int lane) {
  int n = n0 + (lane & 15);
  const _Float16* p = M + n * ld + k0 + ((lane >> 4) << 3);
  return ld16pair(p, p + 16);
}

__device__ __forceinline__ v8f wmma_f16(v16h a, v16h b, v8f c) {
  return __builtin_amdgcn_wmma_f32_16x16x32_f16(false, a, false, b, (short)0, c,
                                                false, false);
}

// butterfly reductions within each 16-lane half-wave (matches C-layout rows)
__device__ __forceinline__ float rmax16(float v) {
  v = fmaxf(v, __shfl_xor(v, 1));
  v = fmaxf(v, __shfl_xor(v, 2));
  v = fmaxf(v, __shfl_xor(v, 4));
  v = fmaxf(v, __shfl_xor(v, 8));
  return v;
}
__device__ __forceinline__ float rsum16(float v) {
  v += __shfl_xor(v, 1);
  v += __shfl_xor(v, 2);
  v += __shfl_xor(v, 4);
  v += __shfl_xor(v, 8);
  return v;
}

// ---------------------------------------------------------------------------
// Kernel 1: fused QKV projection (f32 in -> f16 out), 64 rows per block.
// q16,k16: [B,NH,SEQ,HD] ; vT: [B,NH,HD,SEQ] (transposed for PV B-operand)
// ---------------------------------------------------------------------------
__global__ __launch_bounds__(128) void qkv_proj_kernel(
    const float* __restrict__ hs, const float* __restrict__ qx,
    const float* __restrict__ Wq, const float* __restrict__ bq,
    const float* __restrict__ Wk, const float* __restrict__ bk,
    const float* __restrict__ Wv, const float* __restrict__ bv,
    _Float16* __restrict__ q16, _Float16* __restrict__ k16,
    _Float16* __restrict__ vT) {
  __shared__ __align__(16) _Float16 Xq[64 * 64];
  __shared__ __align__(16) _Float16 Xh[64 * 64];
  __shared__ __align__(16) _Float16 Wsh[3][64 * 64];

  const int t = threadIdx.x, lane = t & 31, w = t >> 5;
  const int row0 = blockIdx.x * 64;

  for (int i = t; i < 4096; i += 128) {
    Xq[i]     = (_Float16)qx[(size_t)row0 * HID + i];
    Xh[i]     = (_Float16)hs[(size_t)row0 * HID + i];
    Wsh[0][i] = (_Float16)Wq[i];
    Wsh[1][i] = (_Float16)Wk[i];
    Wsh[2][i] = (_Float16)Wv[i];
  }
  __syncthreads();

  for (int p = 0; p < 3; ++p) {
    const _Float16* X   = (p == 1) ? Xh : Xq;
    const float* bias   = (p == 0) ? bq : (p == 1) ? bk : bv;
    #pragma unroll
    for (int nt = 0; nt < 4; ++nt) {
      v8f acc = {};
      #pragma unroll
      for (int kk = 0; kk < 2; ++kk)
        acc = wmma_f16(load_A(X, 64, w * 16, kk * 32, lane),
                       load_Bt(Wsh[p], 64, nt * 16, kk * 32, lane), acc);
      const int col = nt * 16 + (lane & 15);
      const float bb = bias[col];
      const int hh = col >> 5, d = col & 31;
      #pragma unroll
      for (int r = 0; r < 8; ++r) {
        const int g = row0 + w * 16 + r + ((lane >> 4) << 3);
        const int b_ = g >> 11, lrow = g & (SEQ - 1);
        const _Float16 val = (_Float16)(acc[r] + bb);
        if (p == 2) {
          vT[(((size_t)b_ * NH + hh) * HD + d) * SEQ + lrow] = val;
        } else {
          const size_t o = (((size_t)b_ * NH + hh) * SEQ + lrow) * HD + d;
          if (p == 0) q16[o] = val; else k16[o] = val;
        }
      }
    }
  }
}

// ---------------------------------------------------------------------------
// Kernel 2: attention core. One block = (b, h, 64-query tile), 4 waves.
// Pass 1: online row max/sum (stats only). Pass 2: recompute S, write
// normalized probs (non-temporal), accumulate ctx = P @ V via WMMA.
// ---------------------------------------------------------------------------
__global__ __launch_bounds__(128) void attn_kernel(
    const _Float16* __restrict__ q16, const _Float16* __restrict__ k16,
    const _Float16* __restrict__ vT, _Float16* __restrict__ ctx16,
    float* __restrict__ probs) {
  __shared__ __align__(16) _Float16 Qs[QTILE * HD];
  __shared__ __align__(16) _Float16 Ks[KTILE * HD];
  __shared__ __align__(16) _Float16 Vs[HD * KTILE];   // [d][key]
  __shared__ __align__(16) _Float16 Ps[4][16 * KTILE];

  const int t = threadIdx.x, lane = t & 31, w = t >> 5;
  const int qt = blockIdx.x & 31;
  const int h  = (blockIdx.x >> 5) & 1;
  const int b  = blockIdx.x >> 6;
  const int q0 = qt * QTILE;
  const size_t bh = (size_t)b * NH + h;

  const _Float16* Qg = q16 + bh * (size_t)SEQ * HD + (size_t)q0 * HD;
  const _Float16* Kg = k16 + bh * (size_t)SEQ * HD;
  const _Float16* Vg = vT  + bh * (size_t)HD * SEQ;

  // stage Q tile (2048 halves = 256 x 16B)
  for (int i = t; i < (QTILE * HD) / 8; i += 128)
    cp16_lds(Qs + i * 8, Qg + i * 8);
  wait_lds_cp();
  __syncthreads();

  const float scale = 0.17677669529663687f;  // 1/sqrt(32)
  const v16h Aq = load_A(Qs, HD, w * 16, 0, lane);  // invariant A operand

  float m[8], l[8];
  #pragma unroll
  for (int r = 0; r < 8; ++r) { m[r] = -1e30f; l[r] = 0.f; }

  // ---------------- pass 1: softmax statistics ----------------
  for (int kt = 0; kt < SEQ; kt += KTILE) {
    __syncthreads();
    // stage K tile (4096 halves = 512 x 16B)
    for (int i = t; i < 512; i += 128)
      cp16_lds(Ks + i * 8, Kg + (size_t)kt * HD + i * 8);
    wait_lds_cp();
    __syncthreads();
    if (kt + KTILE < SEQ)  // pull next tile toward the caches
      __builtin_prefetch(Kg + (size_t)(kt + KTILE) * HD + t * 32, 0, 1);

    v8f S[8];
    #pragma unroll
    for (int nt = 0; nt < 8; ++nt) {
      v8f z = {};
      S[nt] = wmma_f16(Aq, load_Bt(Ks, HD, nt * 16, 0, lane), z);
    }
    #pragma unroll
    for (int r = 0; r < 8; ++r) {
      float mm = S[0][r];
      #pragma unroll
      for (int nt = 1; nt < 8; ++nt) mm = fmaxf(mm, S[nt][r]);
      const float tmax = rmax16(mm * scale);
      const float mn   = fmaxf(m[r], tmax);
      const float corr = __expf(m[r] - mn);
      float s = 0.f;
      #pragma unroll
      for (int nt = 0; nt < 8; ++nt) s += __expf(S[nt][r] * scale - mn);
      l[r] = l[r] * corr + s;   // per-lane partial; m uniform across 16-group
      m[r] = mn;
    }
  }
  float inv_l[8];
  #pragma unroll
  for (int r = 0; r < 8; ++r) inv_l[r] = 1.0f / rsum16(l[r]);

  // ---------------- pass 2: probs + ctx ----------------
  v8f C0 = {}, C1 = {};
  for (int kt = 0; kt < SEQ; kt += KTILE) {
    __syncthreads();
    // stage K + V tiles (512 x 16B each)
    for (int i = t; i < 512; i += 128) {
      cp16_lds(Ks + i * 8, Kg + (size_t)kt * HD + i * 8);
      const int d = i >> 4, seg = i & 15;  // V row d, 16B segment
      cp16_lds(Vs + i * 8, Vg + (size_t)d * SEQ + kt + seg * 8);
    }
    wait_lds_cp();
    __syncthreads();
    if (kt + KTILE < SEQ) {
      __builtin_prefetch(Kg + (size_t)(kt + KTILE) * HD + t * 32, 0, 1);
      __builtin_prefetch(Vg + (size_t)(t & 31) * SEQ + kt + KTILE + (t >> 5) * 32, 0, 1);
    }

    _Float16* Pw = Ps[w];
    #pragma unroll
    for (int nt = 0; nt < 8; ++nt) {
      v8f z = {};
      v8f S = wmma_f16(Aq, load_Bt(Ks, HD, nt * 16, 0, lane), z);
      const int col = nt * 16 + (lane & 15);
      #pragma unroll
      for (int r = 0; r < 8; ++r) {
        const float pv = __expf(S[r] * scale - m[r]) * inv_l[r];
        const int row = r + ((lane >> 4) << 3);
        // write-once 268MB stream: non-temporal, don't pollute L2
        __builtin_nontemporal_store(
            pv, &probs[(bh * (size_t)SEQ + q0 + w * 16 + row) * SEQ + kt + col]);
        Pw[row * KTILE + col] = (_Float16)pv;
      }
    }
    // ctx += P(16xKTILE) @ V(KTILEx32)
    #pragma unroll
    for (int ks = 0; ks < 4; ++ks) {
      const v16h Ap = load_A(Pw, KTILE, 0, ks * 32, lane);
      C0 = wmma_f16(Ap, load_Bt(Vs, KTILE, 0,  ks * 32, lane), C0);
      C1 = wmma_f16(Ap, load_Bt(Vs, KTILE, 16, ks * 32, lane), C1);
    }
  }

  // write ctx as [B, SEQ, HID] f16 (head h occupies cols h*32..h*32+31)
  const int col = lane & 15;
  #pragma unroll
  for (int r = 0; r < 8; ++r) {
    const int row = q0 + w * 16 + r + ((lane >> 4) << 3);
    const size_t base = ((size_t)b * SEQ + row) * HID + h * HD;
    ctx16[base + col]      = (_Float16)C0[r];
    ctx16[base + 16 + col] = (_Float16)C1[r];
  }
}

// ---------------------------------------------------------------------------
// Kernel 3: out = ctx @ Wo^T + bo (f32 output)
// ---------------------------------------------------------------------------
__global__ __launch_bounds__(128) void oproj_kernel(
    const _Float16* __restrict__ ctx16, const float* __restrict__ Wo,
    const float* __restrict__ bo, float* __restrict__ out) {
  __shared__ __align__(16) _Float16 Xs[64 * 64];
  __shared__ __align__(16) _Float16 Wos[64 * 64];
  const int t = threadIdx.x, lane = t & 31, w = t >> 5;
  const int row0 = blockIdx.x * 64;

  for (int i = t; i < 4096; i += 128) {
    Xs[i]  = ctx16[(size_t)row0 * HID + i];
    Wos[i] = (_Float16)Wo[i];
  }
  __syncthreads();

  #pragma unroll
  for (int nt = 0; nt < 4; ++nt) {
    v8f acc = {};
    #pragma unroll
    for (int kk = 0; kk < 2; ++kk)
      acc = wmma_f16(load_A(Xs, 64, w * 16, kk * 32, lane),
                     load_Bt(Wos, 64, nt * 16, kk * 32, lane), acc);
    const int col = nt * 16 + (lane & 15);
    const float bb = bo[col];
    #pragma unroll
    for (int r = 0; r < 8; ++r) {
      const int g = row0 + w * 16 + r + ((lane >> 4) << 3);
      out[(size_t)g * HID + col] = acc[r] + bb;
    }
  }
}

// ---------------------------------------------------------------------------
extern "C" void kernel_launch(void* const* d_in, const int* in_sizes, int n_in,
                              void* d_out, int out_size, void* d_ws, size_t ws_size,
                              hipStream_t stream) {
  const float* hs = (const float*)d_in[0];
  const float* qx = (const float*)d_in[1];
  const float* Wq = (const float*)d_in[2];
  const float* bq = (const float*)d_in[3];
  const float* Wk = (const float*)d_in[4];
  const float* bk = (const float*)d_in[5];
  const float* Wv = (const float*)d_in[6];
  const float* bv = (const float*)d_in[7];
  const float* Wo = (const float*)d_in[8];
  const float* bo = (const float*)d_in[9];

  float* out   = (float*)d_out;
  float* probs = out + (size_t)BATCH * SEQ * HID;

  const size_t NQ = (size_t)BATCH * NH * SEQ * HD;  // 1M halves each
  _Float16* q16 = (_Float16*)d_ws;
  _Float16* k16 = q16 + NQ;
  _Float16* vT  = k16 + NQ;
  _Float16* ctx = vT + NQ;

  qkv_proj_kernel<<<(BATCH * SEQ) / 64, 128, 0, stream>>>(
      hs, qx, Wq, bq, Wk, bk, Wv, bv, q16, k16, vT);
  attn_kernel<<<BATCH * NH * (SEQ / QTILE), 128, 0, stream>>>(
      q16, k16, vT, ctx, probs);
  oproj_kernel<<<(BATCH * SEQ) / 64, 128, 0, stream>>>(ctx, Wo, bo, out);
}
